// AuxModel_63556926046607
// MI455X (gfx1250) — matrix operand order repaired and checked by reading, not compiled, not run on hardware
//
#include <hip/hip_runtime.h>
#include <math.h>

// Model dims
#define B_  2
#define S_  2048
#define D_  512
#define H_  8
#define HD_ 64

typedef __attribute__((ext_vector_type(16))) __bf16 v16bf;
typedef __attribute__((ext_vector_type(8)))  __bf16 v8bf;
typedef __attribute__((ext_vector_type(8)))  float  v8f;

// A-operand (16-bit, 16x32) per-lane element -> K index (as loaded in this file):
//   K = hf*16 + e   -> 16 contiguous K values per lane (same order as B, not the ISA table order).
// B-operand (16-bit, 32x16): same K = hf*16 + e order as A (this file's choice) -> 16 contiguous K rows per lane.

__device__ __forceinline__ v8f v8f_zero() {
  v8f z = {0.f, 0.f, 0.f, 0.f, 0.f, 0.f, 0.f, 0.f};
  return z;
}
__device__ __forceinline__ v16bf cat8(v8bf lo, v8bf hi) {
  return __builtin_shufflevector(lo, hi, 0, 1, 2, 3, 4, 5, 6, 7,
                                 8, 9, 10, 11, 12, 13, 14, 15);
}

// ---------------------------------------------------------------------------
// Pack W[K][N] (f32) into WMMA B-operand order, bf16:
// tiles of (32K x 16N); within a tile, lane-major with 16 contiguous elements
// per lane:  Wb[(kt*(N/16)+nt)*512 + lane*16 + e] = W[kt*32 + hf*16 + e][nt*16+lr]
// ---------------------------------------------------------------------------
__global__ void pack_w_kernel(const float* __restrict__ W,
                              __bf16* __restrict__ Wb, int K, int N) {
  int idx = blockIdx.x * blockDim.x + threadIdx.x;
  int total = K * N;
  if (idx >= total) return;
  int e    = idx & 15;
  int lane = (idx >> 4) & 31;
  int tile = idx >> 9;
  int ntiles = N >> 4;
  int kt = tile / ntiles, nt = tile % ntiles;
  int hf = lane >> 4, lr = lane & 15;
  int k = kt * 32 + hf * 16 + e;
  int n = nt * 16 + lr;
  Wb[idx] = (__bf16)W[(size_t)k * N + n];
}

// ---------------------------------------------------------------------------
// GEMM: C[M,N] = A[M,K] @ W[K,N] (+ res).  A: bf16 row-major.  Wb: packed
// bf16 B-operand order.  f32 accumulate.  One wave computes a 16x64 tile.
// M mult of 16, N mult of 64, K mult of 32.
// ---------------------------------------------------------------------------
__global__ __launch_bounds__(32) void gemm_wmma_kernel(
    const __bf16* __restrict__ A, const __bf16* __restrict__ Wb,
    const float* __restrict__ res, float* __restrict__ C,
    int M, int N, int K) {
  const int lane = threadIdx.x;
  const int hf = lane >> 4, lr = lane & 15;
  const int tn = blockIdx.x, tm = blockIdx.y;
  const int ntiles = N >> 4;
  const __bf16* Arow = A + (size_t)(tm * 16 + lr) * K;
  v8f acc[4];
#pragma unroll
  for (int c = 0; c < 4; ++c) acc[c] = v8f_zero();

  for (int k0 = 0; k0 < K; k0 += 32) {
    v16bf a = cat8(*(const v8bf*)(Arow + k0 + (hf << 4)),
                   *(const v8bf*)(Arow + k0 + (hf << 4) + 8));
    if (k0 + 32 < K) __builtin_prefetch(Arow + k0 + 32, 0, 3);
#pragma unroll
    for (int c = 0; c < 4; ++c) {
      const int nt = tn * 4 + c;
      const __bf16* bp =
          Wb + (((size_t)(k0 >> 5) * ntiles + nt) << 9) + lane * 16;
      v16bf b = cat8(*(const v8bf*)bp, *(const v8bf*)(bp + 8));
      acc[c] = __builtin_amdgcn_wmma_f32_16x16x32_bf16(
          false, a, false, b, (short)0, acc[c], false, false);
    }
  }

  if (res) {
#pragma unroll
    for (int c = 0; c < 4; ++c) {
      const int col = tn * 64 + c * 16 + lr;
#pragma unroll
      for (int r = 0; r < 8; ++r) {
        int m = tm * 16 + r + (hf << 3);
        C[(size_t)m * N + col] = acc[c][r] + res[(size_t)m * N + col];
      }
    }
  } else {
#pragma unroll
    for (int c = 0; c < 4; ++c) {
      const int col = tn * 64 + c * 16 + lr;
#pragma unroll
      for (int r = 0; r < 8; ++r) {
        int m = tm * 16 + r + (hf << 3);
        C[(size_t)m * N + col] = acc[c][r];
      }
    }
  }
}

// ---------------------------------------------------------------------------
// RMSNorm over rows of width D_ ; emits bf16 (next GEMM's A operand)
// ---------------------------------------------------------------------------
__global__ void rmsnorm_kernel(const float* __restrict__ x,
                               const float* __restrict__ w,
                               __bf16* __restrict__ y) {
  __shared__ float red[256];
  const int row = blockIdx.x, tid = threadIdx.x;
  const float* xr = x + (size_t)row * D_;
  float s = 0.f;
  for (int i = tid; i < D_; i += 256) { float v = xr[i]; s += v * v; }
  red[tid] = s; __syncthreads();
  for (int o = 128; o > 0; o >>= 1) {
    if (tid < o) red[tid] += red[tid + o];
    __syncthreads();
  }
  float inv = rsqrtf(red[0] / (float)D_ + 1e-6f);
  for (int i = tid; i < D_; i += 256)
    y[(size_t)row * D_ + i] = (__bf16)(xr[i] * inv * w[i]);
}

// ---------------------------------------------------------------------------
// Split QKV (rows x 3D, f32) -> bf16 Q,K (seq,h,s,hd row-major, RoPE applied)
// and bf16 V transposed: Vt[(sh*HD + hd)*SL + s].
// One thread per (seq, s, h, pair-index i in 0..31).
// ---------------------------------------------------------------------------
__global__ void split_rope_kernel(const float* __restrict__ QKV,
                                  __bf16* __restrict__ Q, __bf16* __restrict__ K,
                                  __bf16* __restrict__ Vt, int nseq, int SL) {
  int idx = blockIdx.x * blockDim.x + threadIdx.x;
  int total = nseq * SL * H_ * 32;
  if (idx >= total) return;
  int i = idx & 31;
  int h = (idx >> 5) & 7;
  int rest = idx >> 8;
  int s = rest % SL;
  int seq = rest / SL;
  size_t row = (size_t)(seq * SL + s) * (3 * D_);
  int co = h * HD_ + i;
  float q1 = QKV[row + co],            q2 = QKV[row + co + 32];
  float k1 = QKV[row + D_ + co],       k2 = QKV[row + D_ + co + 32];
  float v1 = QKV[row + 2 * D_ + co],   v2 = QKV[row + 2 * D_ + co + 32];
  float invf = powf(10000.0f, -(float)i / 32.0f);
  float ang = (float)s * invf;
  float cs = cosf(ang), sn = sinf(ang);
  int sh = seq * H_ + h;
  size_t ob = ((size_t)sh * SL + s) * HD_;
  Q[ob + i]      = (__bf16)( q1 * cs + q2 * sn);
  Q[ob + i + 32] = (__bf16)(-q1 * sn + q2 * cs);
  K[ob + i]      = (__bf16)( k1 * cs + k2 * sn);
  K[ob + i + 32] = (__bf16)(-k1 * sn + k2 * cs);
  Vt[((size_t)sh * HD_ + i)      * SL + s] = (__bf16)v1;
  Vt[((size_t)sh * HD_ + i + 32) * SL + s] = (__bf16)v2;
}

// ---------------------------------------------------------------------------
// Flash attention, one wave per (seq,h, 16-query tile). HD=64.
// maskmode: 0 = dense (router), 1 = block-causal(64) + doc-equality (encoder)
// Inputs bf16 (Q,K row-major; V hd-major).  Output bf16 into (seq*SL+s, D).
// ---------------------------------------------------------------------------
__global__ __launch_bounds__(32) void attn_wmma_kernel(
    const __bf16* __restrict__ Q, const __bf16* __restrict__ Kd,
    const __bf16* __restrict__ Vt, __bf16* __restrict__ O,
    const int* __restrict__ doc, int nseq, int SL, int maskmode) {
  __shared__ __bf16 Plb[16][40];             // 80B row stride: 16B aligned
  const int lane = threadIdx.x;
  const int hf = lane >> 4, lr = lane & 15;
  const int q0 = blockIdx.x * 16;
  const int sh = blockIdx.y;                  // seq*H + h
  const int seq = sh / H_, h = sh % H_;
  const float scale = 0.125f;                 // 1/sqrt(HD)
  const size_t base = (size_t)sh * SL * HD_;

  // Q tile in A-operand layout (two K=32 chunks over HD=64)
  v16bf aQ0, aQ1;
  {
    int qr = q0 + lr; if (qr >= SL) qr = SL - 1;
    const __bf16* qp = Q + base + (size_t)qr * HD_;
    aQ0 = cat8(*(const v8bf*)(qp + (hf << 4)),
               *(const v8bf*)(qp + (hf << 4) + 8));
    aQ1 = cat8(*(const v8bf*)(qp + 32 + (hf << 4)),
               *(const v8bf*)(qp + 32 + (hf << 4) + 8));
  }
  int docq[8];
  if (maskmode == 1) {
#pragma unroll
    for (int r = 0; r < 8; ++r) {
      int m = q0 + r + (hf << 3); if (m >= SL) m = SL - 1;
      docq[r] = doc[(size_t)seq * SL + m];
    }
  }

  float mrow[8], lrow[8];
  v8f oacc[4];
#pragma unroll
  for (int r = 0; r < 8; ++r) { mrow[r] = -3.0e38f; lrow[r] = 0.f; }
#pragma unroll
  for (int t = 0; t < 4; ++t) oacc[t] = v8f_zero();

  int kend = (maskmode == 1) ? (((q0 >> 6) + 1) << 6) : SL;

  for (int k0 = 0; k0 < kend; k0 += 32) {
    // ---- scores: two 16-key tiles, each = 2 WMMAs over HD=64 ----
    v8f sc[2];
    int keyl[2], dk[2];
#pragma unroll
    for (int nb = 0; nb < 2; ++nb) {
      int key = k0 + nb * 16 + lr;
      keyl[nb] = key;
      int kc = key < SL ? key : SL - 1;
      if (maskmode == 1) dk[nb] = doc[(size_t)seq * SL + kc]; else dk[nb] = 0;
      const __bf16* kp = Kd + base + (size_t)kc * HD_;
      v16bf b0 = cat8(*(const v8bf*)(kp + (hf << 4)),
                      *(const v8bf*)(kp + (hf << 4) + 8));
      v16bf b1 = cat8(*(const v8bf*)(kp + 32 + (hf << 4)),
                      *(const v8bf*)(kp + 32 + (hf << 4) + 8));
      v8f s = v8f_zero();
      s = __builtin_amdgcn_wmma_f32_16x16x32_bf16(false, aQ0, false, b0, (short)0, s, false, false);
      s = __builtin_amdgcn_wmma_f32_16x16x32_bf16(false, aQ1, false, b1, (short)0, s, false, false);
      sc[nb] = s;
    }
    // ---- mask, online softmax (row = r + 8*hf, cols striped over 16 lanes) ----
    float p0[8], p1[8];
#pragma unroll
    for (int r = 0; r < 8; ++r) {
      float v0 = sc[0][r] * scale;
      float v1 = sc[1][r] * scale;
      if (keyl[0] >= SL) v0 = -3.0e38f;
      if (keyl[1] >= SL) v1 = -3.0e38f;
      if (maskmode == 1) {
        if (dk[0] != docq[r]) v0 = -3.0e38f;
        if (dk[1] != docq[r]) v1 = -3.0e38f;
      }
      float t = fmaxf(v0, v1);
      for (int off = 8; off > 0; off >>= 1) t = fmaxf(t, __shfl_xor(t, off, 16));
      float mn = fmaxf(mrow[r], t);
      float alpha = __expf(mrow[r] - mn);
      mrow[r] = mn;
      float e0 = __expf(v0 - mn);
      float e1 = __expf(v1 - mn);
      p0[r] = e0; p1[r] = e1;
      float rs = e0 + e1;
      for (int off = 8; off > 0; off >>= 1) rs += __shfl_xor(rs, off, 16);
      lrow[r] = lrow[r] * alpha + rs;
#pragma unroll
      for (int t4 = 0; t4 < 4; ++t4) oacc[t4][r] = oacc[t4][r] * alpha;
    }
    // ---- P (C-layout) -> LDS (bf16) -> A-layout ----
    __syncthreads();
#pragma unroll
    for (int r = 0; r < 8; ++r) {
      Plb[r + (hf << 3)][lr]      = (__bf16)p0[r];
      Plb[r + (hf << 3)][16 + lr] = (__bf16)p1[r];
    }
    __syncthreads();
    v16bf aP = cat8(*(const v8bf*)&Plb[lr][hf << 4],
                    *(const v8bf*)&Plb[lr][(hf << 4) + 8]);
    // ---- O += P @ V : 4 WMMAs (HD split into 4 tiles of 16) ----
    const bool fullblk = (k0 + 32 <= SL);
#pragma unroll
    for (int t4 = 0; t4 < 4; ++t4) {
      const int hd = t4 * 16 + lr;
      const __bf16* vp = Vt + ((size_t)sh * HD_ + hd) * SL + k0 + (hf << 4);
      v16bf bV;
      if (fullblk) {
        bV = cat8(*(const v8bf*)vp, *(const v8bf*)(vp + 8));
      } else {
#pragma unroll
        for (int e = 0; e < 16; ++e) {
          int key = k0 + (hf << 4) + e;
          bV[e] = (key < SL) ? vp[e] : (__bf16)0.f;   // P weight is 0 there
        }
      }
      oacc[t4] = __builtin_amdgcn_wmma_f32_16x16x32_bf16(
          false, aP, false, bV, (short)0, oacc[t4], false, false);
    }
  }

  // ---- normalize and store (bf16) to (seq*SL + m, D) at column h*64+hd ----
#pragma unroll
  for (int r = 0; r < 8; ++r) {
    int m = q0 + r + (hf << 3);
    if (m >= SL) continue;
    float inv = 1.0f / fmaxf(lrow[r], 1e-30f);
#pragma unroll
    for (int t4 = 0; t4 < 4; ++t4) {
      O[((size_t)(seq * SL + m)) * D_ + h * HD_ + t4 * 16 + lr] =
          (__bf16)(oacc[t4][r] * inv);
    }
  }
}

// ---------------------------------------------------------------------------
// SiLU gate: G[rows,1024] = silu(U[:, :1024]) * U[:, 1024:], bf16 out
// ---------------------------------------------------------------------------
__global__ void silu_gate_kernel(const float* __restrict__ U,
                                 __bf16* __restrict__ G, int rows) {
  int idx = blockIdx.x * blockDim.x + threadIdx.x;
  int total = rows * 1024;
  if (idx >= total) return;
  int r = idx >> 10, j = idx & 1023;
  float h1 = U[(size_t)r * 2048 + j];
  float h2 = U[(size_t)r * 2048 + 1024 + j];
  G[idx] = (__bf16)(h1 * (1.0f / (1.0f + __expf(-h1))) * h2);
}

// ---------------------------------------------------------------------------
// Build router sequences: 64 seqs x (64 block tokens + 8 router tokens) x D
// ---------------------------------------------------------------------------
__global__ void assemble_xr_kernel(const float* __restrict__ X,
                                   const float* __restrict__ rt,
                                   float* __restrict__ XR) {
  int idx = blockIdx.x * blockDim.x + threadIdx.x;
  int total = 64 * 72 * D_;
  if (idx >= total) return;
  int d = idx % D_;
  int row = idx / D_;
  int seq = row / 72, p = row % 72;
  XR[idx] = (p < 64) ? X[((size_t)(seq * 64 + p)) * D_ + d]
                     : rt[(size_t)(p - 64) * D_ + d];
}

// Gather the 8 router-token rows of each sequence -> (512, D) bf16
__global__ void gather_rt_kernel(const float* __restrict__ XR,
                                 __bf16* __restrict__ G) {
  int idx = blockIdx.x * blockDim.x + threadIdx.x;
  int total = 512 * D_;
  if (idx >= total) return;
  int d = idx % D_, row = idx / D_;
  int seq = row >> 3, r = row & 7;
  G[idx] = (__bf16)XR[((size_t)(seq * 72 + 64 + r)) * D_ + d];
}

// ---------------------------------------------------------------------------
// Routing head: per (g, bb) compute 16 cosine logits for router & gate keys,
// then top-2 of router logits; gather gate logits at same indices.
// ---------------------------------------------------------------------------
__global__ __launch_bounds__(32) void route_topk_kernel(
    const float* __restrict__ OUT, const float* __restrict__ KR,
    const float* __restrict__ KG, float* __restrict__ out) {
  __shared__ float lrs[16], lgs[16];
  int g = blockIdx.x >> 6;
  int bb = blockIdx.x & 63;
  int b = bb >> 5, l = bb & 31;
  int lsrc = (l + 31) & 31;       // roll by +1 along the block axis
  int r = g >> 2;                 // repeat RM=4
  const float* xrow = OUT + ((size_t)((b * 32 + lsrc) * 8 + r)) * 256;
  int t = threadIdx.x;
  if (t < 16) {
    float dr = 0.f, dg = 0.f, nkr = 0.f, nkg = 0.f, nxr = 0.f, nxg = 0.f;
    for (int d = 0; d < 128; ++d) {
      float xr_ = xrow[d], xg_ = xrow[128 + d];
      float kr_ = KR[((size_t)g * 128 + d) * 16 + t];
      float kg_ = KG[((size_t)g * 128 + d) * 16 + t];
      dr += xr_ * kr_; dg += xg_ * kg_;
      nkr += kr_ * kr_; nkg += kg_ * kg_;
      nxr += xr_ * xr_; nxg += xg_ * xg_;
    }
    lrs[t] = dr / (fmaxf(sqrtf(nxr), 1e-12f) * fmaxf(sqrtf(nkr), 1e-12f));
    lgs[t] = dg / (fmaxf(sqrtf(nxg), 1e-12f) * fmaxf(sqrtf(nkg), 1e-12f));
  }
  __syncthreads();
  if (t == 0) {
    int i0 = 0; float v0 = lrs[0];
    for (int e = 1; e < 16; ++e) if (lrs[e] > v0) { v0 = lrs[e]; i0 = e; }
    int i1 = (i0 == 0) ? 1 : 0; float v1 = lrs[i1];
    for (int e = 0; e < 16; ++e) {
      if (e == i0) continue;
      if (lrs[e] > v1) { v1 = lrs[e]; i1 = e; }
    }
    size_t p = ((size_t)g * 64 + bb) * 2;
    out[p]            = v0;
    out[p + 1]        = v1;
    out[4096 + p]     = lgs[i0];
    out[4096 + p + 1] = lgs[i1];
  }
}

// ---------------------------------------------------------------------------
// Host-side: one full transformer layer
// ---------------------------------------------------------------------------
static void run_layer(const float* qkv_w, const float* o_w, const float* up_w,
                      const float* down_w, const float* n1, const float* n2,
                      float* X, const int* doc, int nseq, int SL, int maskmode,
                      __bf16* XNb, float* QKV, __bf16* Qb, __bf16* Kb,
                      __bf16* Vtb, __bf16* ATTb, float* UP, __bf16* GTb,
                      __bf16* WB, hipStream_t stream) {
  const int rows = nseq * SL;
  rmsnorm_kernel<<<rows, 256, 0, stream>>>(X, n1, XNb);
  pack_w_kernel<<<(512 * 1536 + 255) / 256, 256, 0, stream>>>(qkv_w, WB, 512, 1536);
  gemm_wmma_kernel<<<dim3(1536 / 64, rows / 16), 32, 0, stream>>>(
      XNb, WB, nullptr, QKV, rows, 1536, 512);
  int tot = rows * H_ * 32;
  split_rope_kernel<<<(tot + 255) / 256, 256, 0, stream>>>(QKV, Qb, Kb, Vtb, nseq, SL);
  int qtiles = (SL + 15) / 16;
  attn_wmma_kernel<<<dim3(qtiles, nseq * H_), 32, 0, stream>>>(
      Qb, Kb, Vtb, ATTb, doc, nseq, SL, maskmode);
  pack_w_kernel<<<(512 * 512 + 255) / 256, 256, 0, stream>>>(o_w, WB, 512, 512);
  gemm_wmma_kernel<<<dim3(512 / 64, rows / 16), 32, 0, stream>>>(
      ATTb, WB, X, X, rows, 512, 512);                         // x += o @ o_w
  rmsnorm_kernel<<<rows, 256, 0, stream>>>(X, n2, XNb);
  pack_w_kernel<<<(512 * 2048 + 255) / 256, 256, 0, stream>>>(up_w, WB, 512, 2048);
  gemm_wmma_kernel<<<dim3(2048 / 64, rows / 16), 32, 0, stream>>>(
      XNb, WB, nullptr, UP, rows, 2048, 512);
  int tg = rows * 1024;
  silu_gate_kernel<<<(tg + 255) / 256, 256, 0, stream>>>(UP, GTb, rows);
  pack_w_kernel<<<(1024 * 512 + 255) / 256, 256, 0, stream>>>(down_w, WB, 1024, 512);
  gemm_wmma_kernel<<<dim3(512 / 64, rows / 16), 32, 0, stream>>>(
      GTb, WB, X, X, rows, 512, 1024);                         // x += g @ down
}

extern "C" void kernel_launch(void* const* d_in, const int* in_sizes, int n_in,
                              void* d_out, int out_size, void* d_ws, size_t ws_size,
                              hipStream_t stream) {
  (void)in_sizes; (void)n_in; (void)out_size; (void)ws_size;
  const float* x_in   = (const float*)d_in[0];
  const int*   doc    = (const int*)d_in[1];
  const float* rtok   = (const float*)d_in[2];
  const float* out_w  = (const float*)d_in[3];
  const float* keysr  = (const float*)d_in[4];
  const float* keysg  = (const float*)d_in[5];
  const float* eqkv   = (const float*)d_in[6];
  const float* eo     = (const float*)d_in[7];
  const float* eup    = (const float*)d_in[8];
  const float* edown  = (const float*)d_in[9];
  const float* en1    = (const float*)d_in[10];
  const float* en2    = (const float*)d_in[11];
  const float* rqkv   = (const float*)d_in[12];
  const float* ro     = (const float*)d_in[13];
  const float* rup    = (const float*)d_in[14];
  const float* rdown  = (const float*)d_in[15];
  const float* rn1    = (const float*)d_in[16];
  const float* rn2    = (const float*)d_in[17];

  char* wsb = (char*)d_ws;
  size_t off = 0;
  auto allocB = [&](size_t bytes) {
    void* p = wsb + off;
    off += (bytes + 255) & ~(size_t)255;
    return p;
  };
  float*  X    = (float*) allocB((size_t)4096 * 512 * 4);
  float*  XR   = (float*) allocB((size_t)4608 * 512 * 4);
  float*  QKV  = (float*) allocB((size_t)4608 * 1536 * 4);
  float*  UP   = (float*) allocB((size_t)4608 * 2048 * 4);
  float*  OUT  = (float*) allocB((size_t)512 * 256 * 4);
  __bf16* XNb  = (__bf16*)allocB((size_t)4608 * 512 * 2);
  __bf16* Qb   = (__bf16*)allocB((size_t)64 * 8 * 72 * 64 * 2);  // max(enc,rtr)
  __bf16* Kb   = (__bf16*)allocB((size_t)64 * 8 * 72 * 64 * 2);
  __bf16* Vtb  = (__bf16*)allocB((size_t)64 * 8 * 72 * 64 * 2);
  __bf16* ATTb = (__bf16*)allocB((size_t)4608 * 512 * 2);
  __bf16* GTb  = (__bf16*)allocB((size_t)4608 * 1024 * 2);
  __bf16* GATHb= (__bf16*)allocB((size_t)512 * 512 * 2);
  __bf16* WB   = (__bf16*)allocB((size_t)512 * 2048 * 2);       // max weight

  hipMemcpyAsync(X, x_in, sizeof(float) * 4096 * 512,
                 hipMemcpyDeviceToDevice, stream);

  // 4 encoder layers: nseq=B=2, SL=2048, block-causal + doc mask
  for (int i = 0; i < 4; ++i) {
    run_layer(eqkv + (size_t)i * 512 * 1536, eo + (size_t)i * 512 * 512,
              eup + (size_t)i * 512 * 2048, edown + (size_t)i * 1024 * 512,
              en1 + (size_t)i * 512, en2 + (size_t)i * 512,
              X, doc, 2, 2048, 1, XNb, QKV, Qb, Kb, Vtb, ATTb, UP, GTb, WB, stream);
  }

  // Router sequences: 64 seqs x 72 tokens
  int totxr = 64 * 72 * 512;
  assemble_xr_kernel<<<(totxr + 255) / 256, 256, 0, stream>>>(X, rtok, XR);
  for (int i = 0; i < 2; ++i) {
    run_layer(rqkv + (size_t)i * 512 * 1536, ro + (size_t)i * 512 * 512,
              rup + (size_t)i * 512 * 2048, rdown + (size_t)i * 1024 * 512,
              rn1 + (size_t)i * 512, rn2 + (size_t)i * 512,
              XR, nullptr, 64, 72, 0, XNb, QKV, Qb, Kb, Vtb, ATTb, UP, GTb, WB, stream);
  }

  // Router-token rows -> projection -> routing logits + top-2
  int totg = 512 * 512;
  gather_rt_kernel<<<(totg + 255) / 256, 256, 0, stream>>>(XR, GATHb);
  pack_w_kernel<<<(512 * 256 + 255) / 256, 256, 0, stream>>>(out_w, WB, 512, 256);
  gemm_wmma_kernel<<<dim3(256 / 64, 512 / 16), 32, 0, stream>>>(
      GATHb, WB, nullptr, OUT, 512, 256, 512);
  route_topk_kernel<<<32 * 64, 32, 0, stream>>>(OUT, keysr, keysg, (float*)d_out);
}
